// OmegaTransformerEdge_26645977104517
// MI455X (gfx1250) — matrix operand order repaired and checked by reading, not compiled, not run on hardware
//
#include <hip/hip_runtime.h>

// ---------------------------------------------------------------------------
// Types / helpers
// ---------------------------------------------------------------------------
typedef unsigned short u16;
typedef unsigned int   u32;
typedef __attribute__((ext_vector_type(16))) __bf16 v16bf;
typedef __attribute__((ext_vector_type(8)))  float  v8f;

#define DEVFN __device__ __forceinline__

DEVFN float bf2f(u16 x) { u32 u = ((u32)x) << 16; return __builtin_bit_cast(float, u); }
DEVFN u16 f2bf(float f) {
    u32 u = __builtin_bit_cast(u32, f);
    u32 r = (u + 0x7FFFu + ((u >> 16) & 1u)) >> 16;   // round-to-nearest-even
    return (u16)r;
}

struct Frag32 { uint4 lo, hi; };
DEVFN v16bf make_frag(uint4 lo, uint4 hi) { Frag32 f{lo, hi}; return __builtin_bit_cast(v16bf, f); }

// LDS byte offset (addrspace(3) ptrtoint) for async-to-LDS VDST operands.
typedef __attribute__((address_space(3))) u16 u16_lds;
DEVFN u32 lds_addr(u16* p) { return (u32)(unsigned long long)(u16_lds*)p; }

// Issue one 16-byte async HBM->LDS copy (per active lane); ASYNCcnt tracked.
DEVFN void async_copy_b128(u32 lds_off, const u16* gptr) {
    asm volatile("global_load_async_to_lds_b128 %0, %1, off"
                 :: "v"(lds_off), "v"(gptr) : "memory");
}
DEVFN void wait_async0() {
    asm volatile("s_wait_asynccnt 0x0" ::: "memory");
}

// Problem constants (from the reference)
static constexpr int Bsz = 64, Npt = 256, Cin = 16, Dm = 512, Hh = 8, Ln = 8, Ff = 2048, HD = 64;
static constexpr int S  = Npt + 1;   // 257 real sequence length (CLS + N)
static constexpr int SP = 288;       // padded to a multiple of 32 for WMMA tiling

// ---------------------------------------------------------------------------
// Tiled bf16 WMMA GEMM:  C = epilogue( A[M,K] * B[N,K]^T )  (f32 accum)
// Double-buffered async-to-LDS staging: tile k+1 streams HBM->LDS via
// GLOBAL_LOAD_ASYNC_TO_LDS_B128 while WMMAs consume tile k; synchronized
// with s_wait_asynccnt + workgroup barrier.
// ---------------------------------------------------------------------------
enum : int { F_BIAS = 1, F_RELU = 2, F_RES = 4, F_OUTF32 = 8 };

template <int BM, int BN, int WM, int WN, int FLAGS>
__global__ __launch_bounds__((BM / WM) * (BN / WN) * 32)
void gemm_bf16_wmma(const u16* __restrict__ A, const u16* __restrict__ Bm,
                    void* __restrict__ Cv,
                    const float* __restrict__ bias,     // [N], F_BIAS
                    const float* __restrict__ resid,    // residual (same layout as C), F_RES
                    int M, int N, int K,
                    int lda, int ldb, int ldc)
{
    constexpr int BK = 32;
    constexpr int WAVES_M = BM / WM, WAVES_N = BN / WN;
    constexpr int NTH = WAVES_M * WAVES_N * 32;
    constexpr int MT = WM / 16, NT = WN / 16;
    constexpr int LA = (BM * BK) / (NTH * 8);
    constexpr int LB = (BN * BK) / (NTH * 8);
    static_assert(LA * NTH * 8 == BM * BK, "A tile not evenly divisible");
    static_assert(LB * NTH * 8 == BN * BK, "B tile not evenly divisible");

    __shared__ u16 sA[2][BM * BK];
    __shared__ u16 sB[2][BN * BK];

    const int tid  = threadIdx.x;
    const int wave = tid >> 5, lane = tid & 31;
    const int l15  = lane & 15, lh = (lane >> 4) & 1;
    const int wm   = (wave / WAVES_N) * WM;
    const int wn   = (wave % WAVES_N) * WN;
    const int tileM = blockIdx.y * BM;
    const int tileN = blockIdx.x * BN;

    // Async-stage one BKxBM/BN tile pair into LDS buffer `buf`.
    // All launches in this file have M%BM==0 && N%BN==0; the clamp only keeps
    // addresses legal in the general case (clamped rows feed guarded outputs).
    auto issueTile = [&](int buf, int k0) {
#pragma unroll
        for (int i = 0; i < LA; i++) {
            int linear = (tid + i * NTH) * 8;
            int r = linear / BK, kk = linear % BK;
            int mg = tileM + r; if (mg >= M) mg = M - 1;
            async_copy_b128(lds_addr(&sA[buf][r * BK + kk]),
                            A + (long)mg * lda + k0 + kk);
        }
#pragma unroll
        for (int i = 0; i < LB; i++) {
            int linear = (tid + i * NTH) * 8;
            int r = linear / BK, kk = linear % BK;
            int ng = tileN + r; if (ng >= N) ng = N - 1;
            async_copy_b128(lds_addr(&sB[buf][r * BK + kk]),
                            Bm + (long)ng * ldb + k0 + kk);
        }
    };

    const v8f vzero = {0.f, 0.f, 0.f, 0.f, 0.f, 0.f, 0.f, 0.f};
    v8f acc[MT][NT];
#pragma unroll
    for (int mi = 0; mi < MT; mi++)
#pragma unroll
        for (int ni = 0; ni < NT; ni++) acc[mi][ni] = vzero;

    const int nK = K / BK;
    issueTile(0, 0);
    wait_async0();
    __syncthreads();

    for (int kt = 0; kt < nK; ++kt) {
        const int cur = kt & 1;
        if (kt + 1 < nK) issueTile(cur ^ 1, (kt + 1) * BK);   // overlap with WMMAs

        // A fragment: lane holds row M=l&15; elems 0..7 -> K=8*(l>>4)+0..7,
        // elems 8..15 -> K=16+8*(l>>4)+0..7 (documented CDNA5 layout)
        v16bf fa[MT], fb[NT];
#pragma unroll
        for (int mi = 0; mi < MT; mi++) {
            const u16* p = &sA[cur][(wm + mi * 16 + l15) * BK + lh * 8];
            fa[mi] = make_frag(*(const uint4*)p, *(const uint4*)(p + 16));
        }
        // B fragment: lane holds column N=l&15; same K layout as A fragment
#pragma unroll
        for (int ni = 0; ni < NT; ni++) {
            const u16* p = &sB[cur][(wn + ni * 16 + l15) * BK + lh * 8];
            fb[ni] = make_frag(*(const uint4*)p, *(const uint4*)(p + 16));
        }
#pragma unroll
        for (int mi = 0; mi < MT; mi++)
#pragma unroll
            for (int ni = 0; ni < NT; ni++)
                acc[mi][ni] = __builtin_amdgcn_wmma_f32_16x16x32_bf16(
                    false, fa[mi], false, fb[ni], (short)0, acc[mi][ni], false, false);

        wait_async0();      // next tile landed (own waves' copies)
        __syncthreads();    // ...and everyone else's; also fences buffer reuse
    }

    // Epilogue: C/D layout — elem v of v8f: M = 8*(l>>4)+v, N = l&15
    float* Cf = (float*)Cv;
    u16*   Ch = (u16*)Cv;
#pragma unroll
    for (int mi = 0; mi < MT; mi++) {
#pragma unroll
        for (int ni = 0; ni < NT; ni++) {
#pragma unroll
            for (int v = 0; v < 8; v++) {
                int mg = tileM + wm + mi * 16 + lh * 8 + v;
                int ng = tileN + wn + ni * 16 + l15;
                if (mg < M && ng < N) {
                    float val = acc[mi][ni][v];
                    long cidx = (long)mg * ldc + ng;
                    if (FLAGS & F_BIAS)  val += bias[ng];
                    if (FLAGS & F_RES)   val += resid[cidx];
                    if (FLAGS & F_RELU)  val = fmaxf(val, 0.f);
                    if (FLAGS & F_OUTF32) Cf[cidx] = val;
                    else                  Ch[cidx] = f2bf(val);
                }
            }
        }
    }
}

// ---------------------------------------------------------------------------
// Fused flash attention (bf16 WMMA, online softmax, scores never hit HBM).
// Grid: (SP/96, B*H). Block: 6 waves; each wave owns a 16-row query tile.
// K tile staged via async-to-LDS; V staged transposed through registers.
// ---------------------------------------------------------------------------
__global__ __launch_bounds__(192)
void k_flash_attn(const u16* __restrict__ qkv, const u16* __restrict__ bias,
                  u16* __restrict__ o)
{
    constexpr int KT = 32;                 // kv tile width
    __shared__ u16 sK[KT * HD];            // [n][d]   32 x 64
    __shared__ u16 sV[HD * KT];            // [d][n]   64 x 32 (transposed)
    __shared__ u16 sP[6 * 16 * KT];        // per-wave P transpose patch

    const int z  = blockIdx.y;
    const int b  = z / Hh, hh = z % Hh;
    const long qbase = (long)b * SP * (3 * Dm) + hh * HD;
    const u16* qp = qkv + qbase;           // q rows (row stride 3*Dm)
    const u16* kp = qkv + qbase + Dm;      // k rows
    const u16* vp = qkv + qbase + 2 * Dm;  // v rows
    const u16* bz = bias + (long)z * SP * SP;

    const int tid = threadIdx.x;
    const int wave = tid >> 5, lane = tid & 31;
    const int l15 = lane & 15, lh = (lane >> 4) & 1;
    const int q0 = blockIdx.x * 96 + wave * 16;

    // Q fragments (A layout, 16 rows x 64 d) loaded once, straight from global
    v16bf fq[2];
#pragma unroll
    for (int kf = 0; kf < 2; kf++) {
        const u16* p = qp + (long)(q0 + l15) * (3 * Dm) + kf * 32 + lh * 8;
        fq[kf] = make_frag(*(const uint4*)p, *(const uint4*)(p + 16));
    }

    const v8f vzero = {0.f, 0.f, 0.f, 0.f, 0.f, 0.f, 0.f, 0.f};
    v8f accO[4];
#pragma unroll
    for (int i = 0; i < 4; i++) accO[i] = vzero;
    float mrow[8], lrow[8];
#pragma unroll
    for (int v = 0; v < 8; v++) { mrow[v] = -3.0e38f; lrow[v] = 0.f; }

    u16* myP = &sP[wave * 16 * KT];

    for (int n0 = 0; n0 < SP; n0 += KT) {
        // stage K via async HBM->LDS; V transposed through registers
        for (int i = tid; i < KT * HD / 8; i += 192) {
            int r = i >> 3, dc = (i & 7) * 8;
            async_copy_b128(lds_addr(&sK[r * HD + dc]),
                            kp + (long)(n0 + r) * (3 * Dm) + dc);
            union { uint4 q; u16 s[8]; } u;
            u.q = *(const uint4*)(vp + (long)(n0 + r) * (3 * Dm) + dc);
#pragma unroll
            for (int j = 0; j < 8; j++) sV[(dc + j) * KT + r] = u.s[j];
        }
        wait_async0();
        __syncthreads();

        // S tile: 16 x 32 as two 16x16 C tiles, K-dim = HD in two steps
        v8f accS[2] = {vzero, vzero};
#pragma unroll
        for (int nt = 0; nt < 2; nt++) {
#pragma unroll
            for (int dstep = 0; dstep < 2; dstep++) {
                const u16* p = &sK[(nt * 16 + l15) * HD + dstep * 32 + lh * 8];
                v16bf fk = make_frag(*(const uint4*)p, *(const uint4*)(p + 16));
                accS[nt] = __builtin_amdgcn_wmma_f32_16x16x32_bf16(
                    false, fq[dstep], false, fk, (short)0, accS[nt], false, false);
            }
        }

        // fused scale + edge bias + key-padding mask
        float sv[2][8];
#pragma unroll
        for (int nt = 0; nt < 2; nt++) {
            int n = n0 + nt * 16 + l15;
#pragma unroll
            for (int v = 0; v < 8; v++) {
                int m = q0 + lh * 8 + v;
                float s = accS[nt][v] * 0.125f + bf2f(bz[(long)m * SP + n]);
                sv[nt][v] = (n < S) ? s : -3.0e38f;
            }
        }

        // online softmax (row reductions across the 16-lane column groups)
#pragma unroll
        for (int v = 0; v < 8; v++) {
            float mx = fmaxf(sv[0][v], sv[1][v]);
#pragma unroll
            for (int x = 1; x < 16; x <<= 1) mx = fmaxf(mx, __shfl_xor(mx, x, 32));
            float mnew = fmaxf(mrow[v], mx);
            float alpha = __expf(mrow[v] - mnew);
            float p0 = __expf(sv[0][v] - mnew);
            float p1 = __expf(sv[1][v] - mnew);
            float ps = p0 + p1;
#pragma unroll
            for (int x = 1; x < 16; x <<= 1) ps += __shfl_xor(ps, x, 32);
            lrow[v] = lrow[v] * alpha + ps;
            mrow[v] = mnew;
            sv[0][v] = p0; sv[1][v] = p1;
#pragma unroll
            for (int dt = 0; dt < 4; dt++) accO[dt][v] *= alpha;
        }

        // P: C layout -> per-wave LDS patch [m][n] -> reload in A-frag layout
        // (same-wave LDS accesses are kept in order by DScnt)
#pragma unroll
        for (int nt = 0; nt < 2; nt++)
#pragma unroll
            for (int v = 0; v < 8; v++)
                myP[(lh * 8 + v) * KT + nt * 16 + l15] = f2bf(sv[nt][v]);
        v16bf fp;
        {
            const u16* p = &myP[l15 * KT + lh * 8];
            fp = make_frag(*(const uint4*)p, *(const uint4*)(p + 16));
        }

        // accO += P @ V
#pragma unroll
        for (int dt = 0; dt < 4; dt++) {
            const u16* p = &sV[(dt * 16 + l15) * KT + lh * 8];
            v16bf fv = make_frag(*(const uint4*)p, *(const uint4*)(p + 16));
            accO[dt] = __builtin_amdgcn_wmma_f32_16x16x32_bf16(
                false, fp, false, fv, (short)0, accO[dt], false, false);
        }
        __syncthreads();   // all waves done with sK/sV before next staging
    }

    // normalize and store (padded rows may be non-finite; they are never read)
    const long obase = (long)b * SP * Dm + hh * HD;
#pragma unroll
    for (int v = 0; v < 8; v++) {
        float inv = 1.f / lrow[v];
        int m = q0 + lh * 8 + v;
#pragma unroll
        for (int dt = 0; dt < 4; dt++)
            o[obase + (long)m * Dm + dt * 16 + l15] = f2bf(accO[dt][v] * inv);
    }
}

// ---------------------------------------------------------------------------
// Elementwise / small kernels
// ---------------------------------------------------------------------------
__global__ void k_f32_to_bf16(const float* __restrict__ in, u16* __restrict__ out, long n) {
    long i = (long)blockIdx.x * blockDim.x + threadIdx.x;
    long st = (long)gridDim.x * blockDim.x;
    for (; i < n; i += st) out[i] = f2bf(in[i]);
}

// Edge pair MLP: one block per (b, i); thread t = pair column j.
__global__ __launch_bounds__(256)
void k_edge(const float* __restrict__ x_full, const float* __restrict__ w1,
            const float* __restrict__ b1, const float* __restrict__ w2,
            const float* __restrict__ b2, u16* __restrict__ bias,
            float* __restrict__ central)
{
    int b = blockIdx.x / Npt, i = blockIdx.x % Npt;
    int t = threadIdx.x;
    __shared__ float sw1[48], sb1[16], sw2[Hh * 16], sb2[Hh];
    __shared__ float red[256];
    if (t < 48)      sw1[t] = w1[t];
    if (t < 16)      sb1[t] = b1[t];
    if (t < Hh * 16) sw2[t] = w2[t];
    if (t < Hh)      sb2[t] = b2[t];
    __syncthreads();

    const float PIc = 3.14159265358979323846f;
    float yi = x_full[((long)b * Npt + i) * 16 + 14];
    float pi = x_full[((long)b * Npt + i) * 16 + 15];
    float yj = x_full[((long)b * Npt + t) * 16 + 14];
    float pj = x_full[((long)b * Npt + t) * 16 + 15];
    float dy = fabsf(yi - yj);
    float d  = pi - pj + PIc;
    float m  = fmodf(d, 2.f * PIc); if (m < 0.f) m += 2.f * PIc;
    float dphi = fabsf(m - PIc);
    float dr = sqrtf(dy * dy + dphi * dphi + 1e-8f);

    float hm[16];
#pragma unroll
    for (int k = 0; k < 16; k++) {
        float u = sw1[k * 3 + 0] * dy + sw1[k * 3 + 1] * dphi + sw1[k * 3 + 2] * dr + sb1[k];
        hm[k] = 0.5f * u * (1.f + erff(u * 0.70710678118654752f));   // exact GELU
    }
    long zb = (long)b * Hh;
    for (int hh = 0; hh < Hh; ++hh) {
        float r = sb2[hh];
#pragma unroll
        for (int k = 0; k < 16; k++) r += hm[k] * sw2[hh * 16 + k];
        bias[((zb + hh) * SP + (i + 1)) * (long)SP + (t + 1)] = f2bf(r);
        red[t] = r; __syncthreads();
        for (int s2 = 128; s2 > 0; s2 >>= 1) { if (t < s2) red[t] += red[t + s2]; __syncthreads(); }
        if (t == 0) central[((long)b * Npt + i) * Hh + hh] = red[0] * (1.f / Npt);
        __syncthreads();
    }
}

// Fill CLS row/column of the bias tensor (pad region stays zero from memset).
__global__ __launch_bounds__(256)
void k_cls_bias(const float* __restrict__ central, const float* __restrict__ edge_scale,
                u16* __restrict__ bias)
{
    int z = blockIdx.x, b = z / Hh, hh = z % Hh;
    int q = threadIdx.x;                       // 0..255
    float v = central[((long)b * Npt + q) * Hh + hh] * edge_scale[0];
    u16 hv = f2bf(v);
    long base = (long)z * SP * SP;
    bias[base + (q + 1)]            = hv;      // row 0
    bias[base + (long)(q + 1) * SP] = hv;      // col 0
}

// h[b,0,:]=cls; h[b,1+q,:]=x_node@Wp^T+bp; padded rows zeroed.
__global__ __launch_bounds__(256)
void k_project(const float* __restrict__ x_node, const float* __restrict__ Wp,
               const float* __restrict__ bp, const float* __restrict__ cls,
               float* __restrict__ h)
{
    int row = blockIdx.x, b = row / SP, s = row % SP;
    int t = threadIdx.x;
    __shared__ float xr[Cin];
    if (s >= 1 && s <= Npt && t < Cin) xr[t] = x_node[((long)b * Npt + (s - 1)) * Cin + t];
    __syncthreads();
#pragma unroll
    for (int rep = 0; rep < 2; ++rep) {
        int c = t + rep * 256;
        float v;
        if (s == 0)        v = cls[c];
        else if (s <= Npt) {
            v = bp[c];
#pragma unroll
            for (int k = 0; k < Cin; k++) v += xr[k] * Wp[c * Cin + k];
        } else             v = 0.f;
        h[(long)row * Dm + c] = v;
    }
}

// LayerNorm row -> bf16; padded rows forced to zero.
__global__ __launch_bounds__(256)
void k_layernorm_bf16(const float* __restrict__ h, const float* __restrict__ w,
                      const float* __restrict__ bb, u16* __restrict__ out)
{
    int row = blockIdx.x, s = row % SP;
    int t = threadIdx.x;
    if (s >= S) { out[(long)row * Dm + t] = 0; out[(long)row * Dm + t + 256] = 0; return; }
    const float* x = h + (long)row * Dm;
    float a0 = x[t], a1 = x[t + 256];
    __shared__ float red[256];
    red[t] = a0 + a1; __syncthreads();
    for (int s2 = 128; s2 > 0; s2 >>= 1) { if (t < s2) red[t] += red[t + s2]; __syncthreads(); }
    float mu = red[0] * (1.f / Dm); __syncthreads();
    float d0 = a0 - mu, d1 = a1 - mu;
    red[t] = d0 * d0 + d1 * d1; __syncthreads();
    for (int s2 = 128; s2 > 0; s2 >>= 1) { if (t < s2) red[t] += red[t + s2]; __syncthreads(); }
    float rs = rsqrtf(red[0] * (1.f / Dm) + 1e-5f);
    out[(long)row * Dm + t]       = f2bf(d0 * rs * w[t]       + bb[t]);
    out[(long)row * Dm + t + 256] = f2bf(d1 * rs * w[t + 256] + bb[t + 256]);
}

// Classifier head: LN(h[:,0]) -> GELU(x W1^T + b1) W2^T + b2  -> (B,2)
__global__ __launch_bounds__(256)
void k_head(const float* __restrict__ h, const float* __restrict__ lnw, const float* __restrict__ lnb,
            const float* __restrict__ w1, const float* __restrict__ b1,
            const float* __restrict__ w2, const float* __restrict__ b2,
            float* __restrict__ out)
{
    int b = blockIdx.x, t = threadIdx.x;
    const float* x = h + (long)b * SP * Dm;   // row s=0
    __shared__ float xs[Dm];
    __shared__ float hid[256];
    __shared__ float red[256];
    float a0 = x[t], a1 = x[t + 256];
    red[t] = a0 + a1; __syncthreads();
    for (int s2 = 128; s2 > 0; s2 >>= 1) { if (t < s2) red[t] += red[t + s2]; __syncthreads(); }
    float mu = red[0] * (1.f / Dm); __syncthreads();
    float d0 = a0 - mu, d1 = a1 - mu;
    red[t] = d0 * d0 + d1 * d1; __syncthreads();
    for (int s2 = 128; s2 > 0; s2 >>= 1) { if (t < s2) red[t] += red[t + s2]; __syncthreads(); }
    float rs = rsqrtf(red[0] * (1.f / Dm) + 1e-5f);
    xs[t]       = d0 * rs * lnw[t]       + lnb[t];
    xs[t + 256] = d1 * rs * lnw[t + 256] + lnb[t + 256];
    __syncthreads();
    float u = b1[t];
    for (int k = 0; k < Dm; k++) u += xs[k] * w1[t * Dm + k];
    hid[t] = 0.5f * u * (1.f + erff(u * 0.70710678118654752f));
    __syncthreads();
    if (t < 2) {
        float o = b2[t];
        for (int k = 0; k < 256; k++) o += hid[k] * w2[t * 256 + k];
        out[b * 2 + t] = o;
    }
}

// ---------------------------------------------------------------------------
// Launch
// ---------------------------------------------------------------------------
extern "C" void kernel_launch(void* const* d_in, const int* in_sizes, int n_in,
                              void* d_out, int out_size, void* d_ws, size_t ws_size,
                              hipStream_t stream)
{
    (void)in_sizes; (void)n_in; (void)out_size; (void)ws_size;
    const float* x_node = (const float*)d_in[0];
    const float* x_full = (const float*)d_in[1];
    const float* Wp     = (const float*)d_in[2];
    const float* bp     = (const float*)d_in[3];
    const float* cls    = (const float*)d_in[4];
    const float* e_w1   = (const float*)d_in[5];
    const float* e_b1   = (const float*)d_in[6];
    const float* e_w2   = (const float*)d_in[7];
    const float* e_b2   = (const float*)d_in[8];
    const float* edge_scale = (const float*)d_in[9];
    const float* qkv_w  = (const float*)d_in[10];
    const float* qkv_b  = (const float*)d_in[11];
    const float* out_w  = (const float*)d_in[12];
    const float* out_b  = (const float*)d_in[13];
    const float* ln1_w  = (const float*)d_in[14];
    const float* ln1_b  = (const float*)d_in[15];
    const float* ln2_w  = (const float*)d_in[16];
    const float* ln2_b  = (const float*)d_in[17];
    const float* ff1_w  = (const float*)d_in[18];
    const float* ff1_b  = (const float*)d_in[19];
    const float* ff2_w  = (const float*)d_in[20];
    const float* ff2_b  = (const float*)d_in[21];
    const float* cl_ln_w = (const float*)d_in[22];
    const float* cl_ln_b = (const float*)d_in[23];
    const float* cl_w1  = (const float*)d_in[24];
    const float* cl_b1  = (const float*)d_in[25];
    const float* cl_w2  = (const float*)d_in[26];
    const float* cl_b2  = (const float*)d_in[27];

    // Workspace bump allocator (~350 MB total)
    char* ws = (char*)d_ws;
    size_t off = 0;
    auto alloc = [&](size_t bytes) -> char* {
        char* p = ws + off; off = (off + bytes + 255) & ~(size_t)255; return p;
    };
    const long M    = (long)Bsz * SP;          // 18432 padded rows
    const long SPSP = (long)SP * SP;

    u16*  qkvw_bf = (u16*)alloc((size_t)Ln * 3 * Dm * Dm * 2);
    u16*  outw_bf = (u16*)alloc((size_t)Ln * Dm * Dm * 2);
    u16*  ff1w_bf = (u16*)alloc((size_t)Ln * Ff * Dm * 2);
    u16*  ff2w_bf = (u16*)alloc((size_t)Ln * Dm * Ff * 2);
    float* h      = (float*)alloc((size_t)M * Dm * 4);
    u16*  x_bf    = (u16*)alloc((size_t)M * Dm * 2);
    u16*  qkv_bf  = (u16*)alloc((size_t)M * 3 * Dm * 2);
    u16*  bias_bf = (u16*)alloc((size_t)Bsz * Hh * SPSP * 2);
    u16*  o_bf    = (u16*)alloc((size_t)M * Dm * 2);
    u16*  ff_bf   = (u16*)alloc((size_t)M * Ff * 2);
    float* central = (float*)alloc((size_t)Bsz * Npt * Hh * 4);

    // ---- prep: weight f32->bf16, bias tensor, projection
    hipMemsetAsync(bias_bf, 0, (size_t)Bsz * Hh * SPSP * 2, stream);
    k_f32_to_bf16<<<2048, 256, 0, stream>>>(qkv_w, qkvw_bf, (long)Ln * 3 * Dm * Dm);
    k_f32_to_bf16<<<2048, 256, 0, stream>>>(out_w, outw_bf, (long)Ln * Dm * Dm);
    k_f32_to_bf16<<<2048, 256, 0, stream>>>(ff1_w, ff1w_bf, (long)Ln * Ff * Dm);
    k_f32_to_bf16<<<2048, 256, 0, stream>>>(ff2_w, ff2w_bf, (long)Ln * Dm * Ff);
    k_edge<<<Bsz * Npt, 256, 0, stream>>>(x_full, e_w1, e_b1, e_w2, e_b2, bias_bf, central);
    k_cls_bias<<<Bsz * Hh, 256, 0, stream>>>(central, edge_scale, bias_bf);
    k_project<<<Bsz * SP, 256, 0, stream>>>(x_node, Wp, bp, cls, h);

    // ---- transformer layers
    for (int l = 0; l < Ln; ++l) {
        // LN1 -> x_bf
        k_layernorm_bf16<<<Bsz * SP, 256, 0, stream>>>(h, ln1_w + l * Dm, ln1_b + l * Dm, x_bf);

        // QKV: [M,512] x [1536,512]^T + b -> bf16 [M,1536]
        gemm_bf16_wmma<128, 128, 64, 32, F_BIAS>
            <<<dim3(12, 144, 1), 256, 0, stream>>>(
            x_bf, qkvw_bf + (size_t)l * 3 * Dm * Dm, qkv_bf,
            qkv_b + (size_t)l * 3 * Dm, nullptr,
            (int)M, 3 * Dm, Dm, Dm, Dm, 3 * Dm);

        // fused attention: q,k,v + edge bias -> o (scores stay on-chip)
        k_flash_attn<<<dim3(SP / 96, Bsz * Hh), 192, 0, stream>>>(qkv_bf, bias_bf, o_bf);

        // h += o @ out_w^T + out_b   (f32 out, residual)
        gemm_bf16_wmma<128, 128, 64, 32, F_BIAS | F_RES | F_OUTF32>
            <<<dim3(4, 144, 1), 256, 0, stream>>>(
            o_bf, outw_bf + (size_t)l * Dm * Dm, h,
            out_b + (size_t)l * Dm, h,
            (int)M, Dm, Dm, Dm, Dm, Dm);

        // LN2 -> x_bf
        k_layernorm_bf16<<<Bsz * SP, 256, 0, stream>>>(h, ln2_w + l * Dm, ln2_b + l * Dm, x_bf);

        // FF1: relu(x W1^T + b1) -> bf16 [M, 2048]
        gemm_bf16_wmma<128, 128, 64, 32, F_BIAS | F_RELU>
            <<<dim3(16, 144, 1), 256, 0, stream>>>(
            x_bf, ff1w_bf + (size_t)l * Ff * Dm, ff_bf,
            ff1_b + (size_t)l * Ff, nullptr,
            (int)M, Ff, Dm, Dm, Dm, Ff);

        // h += mid W2^T + b2
        gemm_bf16_wmma<128, 128, 64, 32, F_BIAS | F_RES | F_OUTF32>
            <<<dim3(4, 144, 1), 256, 0, stream>>>(
            ff_bf, ff2w_bf + (size_t)l * Dm * Ff, h,
            ff2_b + (size_t)l * Dm, h,
            (int)M, Dm, Ff, Ff, Ff, Dm);
    }

    k_head<<<Bsz, 256, 0, stream>>>(h, cl_ln_w, cl_ln_b, cl_w1, cl_b1, cl_w2, cl_b2, (float*)d_out);
}
